// Model_84920093377374
// MI455X (gfx1250) — matrix operand order repaired and checked by reading, not compiled, not run on hardware
//
#include <hip/hip_runtime.h>

// ---------------------------------------------------------------------------
// Per-species 4-layer SiLU MLP + segment-sum, fused, BF16 WMMA on gfx1250.
// Weights pre-converted to bf16 (transposed, chunk-tiled) in d_ws; B-chunk
// staging uses gfx1250 async global->LDS loads; B fragments double-buffered.
// ---------------------------------------------------------------------------

typedef __attribute__((ext_vector_type(16))) __bf16 v16bf;
typedef __attribute__((ext_vector_type(2)))  __bf16 v2bf;
typedef __attribute__((ext_vector_type(2)))  float  v2f;
typedef __attribute__((ext_vector_type(8)))  float  v8f;
typedef __attribute__((vector_size(16)))     int    gv4i;   // matches async builtin param

#define NSPEC   4
#define NPER    12500
#define NFEAT   1024
#define HID     256
#define NSTRUCT 500

#define ROWS    64     // atoms per block
#define KCHUNK  32     // K per WMMA step (bf16)
#define A_STR   36     // sA row stride (bf16 elems): 72 B rows, 8 B aligned stores
#define B_STR   40     // sBt row stride (bf16 elems): 80 B rows, 16 B aligned stores
#define H_STR   264    // activation row stride (bf16 elems)

#define WT1_ELEMS ((size_t)NSPEC * HID * NFEAT)           // bf16 elems
#define WT2_ELEMS ((size_t)NSPEC * HID * HID)
#define WT_TOTAL  (WT1_ELEMS + 2 * WT2_ELEMS)             // 1,572,864
#define WS_NEEDED (WT_TOTAL * sizeof(unsigned short))     // 3 MiB

#if defined(__HIP_DEVICE_COMPILE__) && __has_builtin(__builtin_amdgcn_global_load_async_to_lds_b128)
#define HAVE_ASYNC_LDS 1
#else
#define HAVE_ASYNC_LDS 0
#endif

#if defined(__HIP_DEVICE_COMPILE__) && __has_builtin(__builtin_amdgcn_cvt_pk_bf16_f32)
#define HAVE_CVT_PK_BF16 1
#else
#define HAVE_CVT_PK_BF16 0
#endif

union BF16x16 { unsigned u[8]; v16bf v; };

__device__ __forceinline__ unsigned short f2bf(float f) {
#if HAVE_CVT_PK_BF16
    union { v2bf v; unsigned short s[2]; } u;
    u.v = __builtin_amdgcn_cvt_pk_bf16_f32(f, f);
    return u.s[0];
#else
    union { __bf16 b; unsigned short s; } u;
    u.b = (__bf16)f;                       // fptrunc, RNE; native cvt if available
    return u.s;
#endif
}
// pack two floats into one dword of bf16 (lo = a, hi = b)
__device__ __forceinline__ unsigned f2bf_pk(float a, float b) {
#if HAVE_CVT_PK_BF16
    union { v2bf v; unsigned u; } u;
    u.v = __builtin_amdgcn_cvt_pk_bf16_f32(a, b);
    return u.u;
#else
    union { v2bf v; unsigned u; } u;
    v2f in = {a, b};
    u.v = __builtin_convertvector(in, v2bf);
    return u.u;
#endif
}
__device__ __forceinline__ float bf2f(unsigned short b) {
    return __uint_as_float(((unsigned)b) << 16);
}
__device__ __forceinline__ float silu_f(float v) {
    return v * (1.0f / (1.0f + __expf(-v)));
}

template<bool USE_WT>
__device__ __forceinline__ void wait_b_stage() {
#if HAVE_ASYNC_LDS
    if constexpr (USE_WT) {
#if __has_builtin(__builtin_amdgcn_s_wait_asynccnt)
        __builtin_amdgcn_s_wait_asynccnt(0);
#else
        asm volatile("s_wait_asynccnt 0x0" ::: "memory");
#endif
    }
#endif
}

__global__ void zero_out_kernel(float* __restrict__ out, int n) {
    int i = blockIdx.x * blockDim.x + threadIdx.x;
    if (i < n) out[i] = 0.0f;
}

// One-shot: W1/W2/W3 fp32 -> bf16, transposed + chunk-tiled: [s][kc][n][kk]
__global__ __launch_bounds__(256) void convert_weights_kernel(
    const float* __restrict__ W1, const float* __restrict__ W2,
    const float* __restrict__ W3, unsigned short* __restrict__ wt)
{
    size_t i = (size_t)blockIdx.x * 256 + threadIdx.x;
    if (i < WT1_ELEMS) {                                   // 2^20 elems
        size_t kk = i & 31, n = (i >> 5) & 255;
        size_t c  = (i >> 13) & 31, s = i >> 18;
        wt[i] = f2bf(W1[((s * NFEAT) + (c * 32 + kk)) * HID + n]);
        return;
    }
    size_t j = i - WT1_ELEMS;
    if (j < WT2_ELEMS) {
        size_t kk = j & 31, n = (j >> 5) & 255;
        size_t c  = (j >> 13) & 7, s = j >> 16;
        wt[i] = f2bf(W2[((s * HID) + (c * 32 + kk)) * HID + n]);
        return;
    }
    size_t m = j - WT2_ELEMS;
    if (m < WT2_ELEMS) {
        size_t kk = m & 31, n = (m >> 5) & 255;
        size_t c  = (m >> 13) & 7, s = m >> 16;
        wt[i] = f2bf(W3[((s * HID) + (c * 32 + kk)) * HID + n]);
    }
}

template<bool USE_WT>
__global__ __launch_bounds__(256) void mlp_energy_kernel(
    const float* __restrict__ features,
    const float* __restrict__ W1, const float* __restrict__ b1,
    const float* __restrict__ W2, const float* __restrict__ b2,
    const float* __restrict__ W3, const float* __restrict__ b3,
    const float* __restrict__ W4, const float* __restrict__ b4,
    const int*   __restrict__ sidx,
    const unsigned short* __restrict__ wt,
    float* __restrict__ out)
{
    __shared__ unsigned short sA [ROWS * A_STR];   // 64x32 feature chunk (bf16)
    __shared__ unsigned short sBt[HID  * B_STR];   // 32x256 W chunk, N-major (bf16)
    __shared__ unsigned short sH [ROWS * H_STR];   // 64x256 activations (bf16)

    const int tid   = threadIdx.x;
    const int lane  = tid & 31;
    const int wave  = tid >> 5;
    const int strip = wave >> 1;            // 16-row strip: 0..3
    const int colb  = (wave & 1) * 128;     // column half base
    const int l15   = lane & 15;
    const int kg    = lane >> 4;            // lane group (0 or 1)

    const int sp = blockIdx.y;
    const int r0 = blockIdx.x * ROWS;

    const float* F  = features + (size_t)sp * NPER * NFEAT;
    const float* w1 = W1 + (size_t)sp * NFEAT * HID;
    const float* w2 = W2 + (size_t)sp * HID * HID;
    const float* w3 = W3 + (size_t)sp * HID * HID;
    const float* w4 = W4 + (size_t)sp * HID;

    const unsigned short* wt1s = wt + (size_t)sp * HID * NFEAT;
    const unsigned short* wt2s = wt + WT1_ELEMS + (size_t)sp * HID * HID;
    const unsigned short* wt3s = wt2s + WT2_ELEMS;

    // Stage one 32xK B-chunk (all 256 cols, N-major) into sBt.
    auto stage_B = [&](const float* __restrict__ Wf,
                       const unsigned short* __restrict__ Wt, int kc) {
        if constexpr (USE_WT) {
            // tiled bf16: chunk (kc/32) is 16 KB contiguous; thread t = row n=t
            const unsigned short* src = Wt + ((size_t)(kc >> 5) * HID + tid) * KCHUNK;
            unsigned short* dst = &sBt[tid * B_STR];
#if HAVE_ASYNC_LDS
            __builtin_amdgcn_global_load_async_to_lds_b128((gv4i*)src, (gv4i*)dst,  0, 0);
            __builtin_amdgcn_global_load_async_to_lds_b128((gv4i*)src, (gv4i*)dst, 16, 0);
            __builtin_amdgcn_global_load_async_to_lds_b128((gv4i*)src, (gv4i*)dst, 32, 0);
            __builtin_amdgcn_global_load_async_to_lds_b128((gv4i*)src, (gv4i*)dst, 48, 0);
#else
            const uint4* s4 = (const uint4*)src;
            uint4* d4 = (uint4*)dst;
            d4[0] = s4[0]; d4[1] = s4[1]; d4[2] = s4[2]; d4[3] = s4[3];
#endif
        } else {
            #pragma unroll 4
            for (int i = 0; i < KCHUNK; ++i)
                sBt[tid * B_STR + i] = f2bf(Wf[(size_t)(kc + i) * HID + tid]);
        }
    };

    // Load one 32x16 B fragment (tile t of this wave's column half) from sBt.
    auto load_bfrag = [&](int t) {
        BF16x16 b;
        int n = colb + t * 16 + l15;
        #pragma unroll
        for (int r = 0; r < 8; ++r)
            b.u[r] = *(const unsigned*)&sBt[n * B_STR + ((r & 3) << 1) + ((r >> 2) << 4) + (kg << 3)];
        return b;
    };

    // ---------------- Layer 1: [64 x 1024] @ [1024 x 256] -----------------
    v8f acc[8] = {};

    for (int kc = 0; kc < NFEAT; kc += KCHUNK) {
        stage_B(w1, wt1s, kc);
        // stage A chunk: 64 rows x 32 K, fp32 -> bf16 (zero-pad tail rows)
        #pragma unroll
        for (int i = 0; i < 2; ++i) {
            int idx  = tid + i * 256;          // float4 index 0..511
            int row  = idx >> 3;
            int k4   = (idx & 7) << 2;
            int grow = r0 + row;
            float4 f = make_float4(0.f, 0.f, 0.f, 0.f);
            if (grow < NPER)
                f = *(const float4*)(F + (size_t)grow * NFEAT + kc + k4);
            *(uint2*)&sA[row * A_STR + k4] =
                make_uint2(f2bf_pk(f.x, f.y), f2bf_pk(f.z, f.w));
        }
        wait_b_stage<USE_WT>();
        __syncthreads();

        BF16x16 a;
        #pragma unroll
        for (int r = 0; r < 8; ++r) {
            int k = ((r & 3) << 1) + ((r >> 2) << 4) + (kg << 3);
            a.u[r] = *(const unsigned*)&sA[(strip * 16 + l15) * A_STR + k];
        }
        BF16x16 bcur = load_bfrag(0);
        #pragma unroll
        for (int t = 0; t < 8; ++t) {
            BF16x16 bnext = load_bfrag((t + 1) & 7);   // prefetch next tile
            acc[t] = __builtin_amdgcn_wmma_f32_16x16x32_bf16(
                false, a.v, false, bcur.v, (short)0, acc[t], false, false);
            bcur = bnext;
        }
        __syncthreads();
    }
    // epilogue L1: bias + SiLU -> sH (bf16)
    #pragma unroll
    for (int t = 0; t < 8; ++t) {
        int n = colb + t * 16 + l15;
        float bias = b1[sp * HID + n];
        #pragma unroll
        for (int r = 0; r < 8; ++r) {
            int m = strip * 16 + r + (kg << 3);
            sH[m * H_STR + n] = f2bf(silu_f(acc[t][r] + bias));
        }
    }
    __syncthreads();

    // ---------------- Layers 2 & 3: [64 x 256] @ [256 x 256], in-place ----
    auto hidden_layer = [&](const float* __restrict__ Wf,
                            const unsigned short* __restrict__ Wt,
                            const float* __restrict__ bvec) {
        v8f c[8] = {};
        for (int kc = 0; kc < HID; kc += KCHUNK) {
            stage_B(Wf, Wt, kc);
            wait_b_stage<USE_WT>();
            __syncthreads();

            BF16x16 a;
            #pragma unroll
            for (int r = 0; r < 8; ++r) {
                int k = kc + ((r & 3) << 1) + ((r >> 2) << 4) + (kg << 3);
                a.u[r] = *(const unsigned*)&sH[(strip * 16 + l15) * H_STR + k];
            }
            BF16x16 bcur = load_bfrag(0);
            #pragma unroll
            for (int t = 0; t < 8; ++t) {
                BF16x16 bnext = load_bfrag((t + 1) & 7);
                c[t] = __builtin_amdgcn_wmma_f32_16x16x32_bf16(
                    false, a.v, false, bcur.v, (short)0, c[t], false, false);
                bcur = bnext;
            }
            __syncthreads();   // all reads of sH/sBt for this chunk complete
        }
        // in-place epilogue: safe, all K-reads of sH finished at last barrier
        #pragma unroll
        for (int t = 0; t < 8; ++t) {
            int n = colb + t * 16 + l15;
            float bias = bvec[n];
            #pragma unroll
            for (int r = 0; r < 8; ++r) {
                int m = strip * 16 + r + (kg << 3);
                sH[m * H_STR + n] = f2bf(silu_f(c[t][r] + bias));
            }
        }
        __syncthreads();
    };
    hidden_layer(w2, wt2s, b2 + sp * HID);
    hidden_layer(w3, wt3s, b3 + sp * HID);

    // ---------------- Layer 4: per-row dot(h3, W4) + b4, segment add ------
    {
        float* sRed = (float*)sBt;          // reuse staging LDS as fp32 scratch
        int row = tid >> 2;                 // 0..63
        int seg = tid & 3;                  // 4 partial sums per row
        const unsigned short* hrow = sH + row * H_STR;
        float sum = 0.0f;
        #pragma unroll 8
        for (int k = seg * 64; k < seg * 64 + 64; ++k)
            sum += bf2f(hrow[k]) * w4[k];
        sRed[tid] = sum;
        __syncthreads();
        if (seg == 0) {
            float e = sRed[tid] + sRed[tid + 1] + sRed[tid + 2] + sRed[tid + 3]
                      + b4[sp];
            int grow = r0 + row;
            if (grow < NPER)
                atomicAdd(&out[sidx[(size_t)sp * NPER + grow]], e);
        }
    }
}

extern "C" void kernel_launch(void* const* d_in, const int* in_sizes, int n_in,
                              void* d_out, int out_size, void* d_ws, size_t ws_size,
                              hipStream_t stream) {
    const float* features = (const float*)d_in[0];
    const float* W1 = (const float*)d_in[1];
    const float* b1 = (const float*)d_in[2];
    const float* W2 = (const float*)d_in[3];
    const float* b2 = (const float*)d_in[4];
    const float* W3 = (const float*)d_in[5];
    const float* b3 = (const float*)d_in[6];
    const float* W4 = (const float*)d_in[7];
    const float* b4 = (const float*)d_in[8];
    const int*   sidx = (const int*)d_in[9];
    float* out = (float*)d_out;

    zero_out_kernel<<<dim3((NSTRUCT + 255) / 256), dim3(256), 0, stream>>>(out, NSTRUCT);

    dim3 grid((NPER + ROWS - 1) / ROWS, NSPEC);   // 196 x 4 workgroups

    bool use_wt = (d_ws != nullptr) && (ws_size >= WS_NEEDED);
    if (use_wt) {
        unsigned short* wt = (unsigned short*)d_ws;
        convert_weights_kernel<<<dim3((unsigned)(WT_TOTAL / 256)), dim3(256), 0, stream>>>(
            W1, W2, W3, wt);
        mlp_energy_kernel<true><<<grid, dim3(256), 0, stream>>>(
            features, W1, b1, W2, b2, W3, b3, W4, b4, sidx, wt, out);
    } else {
        mlp_energy_kernel<false><<<grid, dim3(256), 0, stream>>>(
            features, W1, b1, W2, b2, W3, b3, W4, b4, sidx, nullptr, out);
    }
}
